// SparseResBlock_3169685865336
// MI455X (gfx1250) — matrix operand order repaired and checked by reading, not compiled, not run on hardware
//
#include <hip/hip_runtime.h>

// ---------- types ----------
typedef __attribute__((ext_vector_type(4)))  unsigned int u32x4;
typedef __attribute__((ext_vector_type(8)))  float        v8f;
typedef __attribute__((ext_vector_type(16))) __bf16       v16bf;
typedef int v4i __attribute__((vector_size(16)));   // matches builtin param type

union V16 { u32x4 u[2]; v16bf v; };

__device__ __host__ inline __bf16 f2bf(float f) { return (__bf16)f; }

// ---------- async global->LDS staging (CDNA5 path), with safe fallback ----------
#if defined(__has_builtin)
#if __has_builtin(__builtin_amdgcn_global_load_async_to_lds_b128)
#define USE_ASYNC_LDS 1
#endif
#endif

__device__ inline void copy_chunk(u32x4* dst_lds, const u32x4* src_glob) {
#ifdef USE_ASYNC_LDS
    __builtin_amdgcn_global_load_async_to_lds_b128(
        (__attribute__((address_space(1))) v4i*)src_glob,
        (__attribute__((address_space(3))) v4i*)dst_lds, 0, 0);
#else
    *dst_lds = *src_glob;
#endif
}

__device__ inline void stage_fence() {
#ifdef USE_ASYNC_LDS
#if __has_builtin(__builtin_amdgcn_s_wait_asynccnt)
    __builtin_amdgcn_s_wait_asynccnt(0);
#else
    asm volatile("s_wait_asynccnt 0x0" ::: "memory");
#endif
#endif
}

// ---------- BN stats: per-block partial sums (deterministic, no atomics) ----------
template<int C>
__global__ void bn_stats_partial(const float* __restrict__ X, int n,
                                 float* __restrict__ psum, float* __restrict__ psq) {
    constexpr int COPIES = 256 / C;
    const int t = threadIdx.x;
    const int c = t % C;
    const int cp = t / C;
    float s = 0.f, q = 0.f;
    for (int r = blockIdx.x * COPIES + cp; r < n; r += gridDim.x * COPIES) {
        float v = X[(size_t)r * C + c];
        s += v; q += v * v;
    }
    __shared__ float ls[256], lq[256];
    ls[t] = s; lq[t] = q;
    __syncthreads();
    for (int off = 128; off >= C; off >>= 1) {
        if (t < off) { ls[t] += ls[t + off]; lq[t] += lq[t + off]; }
        __syncthreads();
    }
    if (t < C) {
        psum[blockIdx.x * C + t] = ls[t];
        psq [blockIdx.x * C + t] = lq[t];
    }
}

// ---------- reduce partials -> per-channel scale/bias ----------
__global__ void bn_params(const float* __restrict__ psum, const float* __restrict__ psq,
                          const float* __restrict__ gamma, const float* __restrict__ beta,
                          int nblk, int C, float inv_n,
                          float* __restrict__ a, float* __restrict__ b) {
    const int t = threadIdx.x;
    if (t >= C) return;
    float s = 0.f, q = 0.f;
    for (int i = 0; i < nblk; ++i) { s += psum[i * C + t]; q += psq[i * C + t]; }
    float mean = s * inv_n;
    float var  = q * inv_n - mean * mean;
    float av   = gamma[t] * rsqrtf(var + 1e-5f);
    a[t] = av;
    b[t] = beta[t] - mean * av;
}

// ---------- y = bf16(relu(x*a+b)); optionally raw = bf16(x) ----------
__global__ void bn_relu_cast(const float* __restrict__ X,
                             const float* __restrict__ a, const float* __restrict__ b,
                             __bf16* __restrict__ Y, __bf16* __restrict__ RAW,
                             int total, int C) {
    const int i = blockIdx.x * 256 + threadIdx.x;
    if (i >= total) return;
    const int c = i % C;
    float x = X[i];
    float y = fmaxf(fmaf(x, a[c], b[c]), 0.f);
    Y[i] = f2bf(y);
    if (RAW) RAW[i] = f2bf(x);
}

// ---------- pack fp32 weights [nk][K][128] into WMMA-B per-lane bf16 layout ----------
// chunk id = (((k*KB + kb)*2 + c)*8 + nt)*32 + lane ; chunk = 8 bf16 = 16 B
// lane holds column n = nt*16 + (lane&15); K = kb*32 + 16*(lane>>4) + 8*c + j
__global__ void pack_weights(const float* __restrict__ W, __bf16* __restrict__ dst,
                             int nk, int KB) {
    const int id = blockIdx.x * 256 + threadIdx.x;
    const int total = nk * KB * 512;
    if (id >= total) return;
    const int ch   = id & 511;
    const int kkb  = id >> 9;
    const int k    = kkb / KB;
    const int kb   = kkb % KB;
    const int lane = ch & 31;
    const int nt   = (ch >> 5) & 7;
    const int c    = ch >> 8;
    const int h    = lane >> 4;
    const int ncol = nt * 16 + (lane & 15);
    const int K    = KB * 32;
    const float* Wk = W + (size_t)k * K * 128;
    __bf16* o = dst + (size_t)id * 8;
    const int kbase = kb * 32 + 16 * h + 8 * c;
#pragma unroll
    for (int j = 0; j < 8; ++j)
        o[j] = f2bf(Wk[(size_t)(kbase + j) * 128 + ncol]);
}

// ---------- GEMM helpers ----------
__device__ inline void gemm_step(const u32x4* ldsB, u32x4 alo, u32x4 ahi,
                                 int lane, v8f acc[8]) {
    V16 a; a.u[0] = alo; a.u[1] = ahi;
#pragma unroll
    for (int nt = 0; nt < 8; ++nt) {
        V16 b;
        b.u[0] = ldsB[nt * 32 + lane];
        b.u[1] = ldsB[256 + nt * 32 + lane];
        acc[nt] = __builtin_amdgcn_wmma_f32_16x16x32_bf16(
            false, a.v, false, b.v, (short)0, acc[nt], false, false);
    }
}

// ---------- submanifold conv via bf16 WMMA ----------
// Block = 256 threads = 8 waves; wave w owns rows [blk*128 + w*16, +16), all 128 cols.
// B (weights) staged per kernel-offset into LDS (async when available): 2 barriers per
// offset instead of per K-step; A gathers issued before the barrier to overlap staging.
template<int CIN, bool SKIP>
__global__ void __launch_bounds__(256)
subm_conv_wmma(const __bf16* __restrict__ X, const __bf16* __restrict__ Wp,
               const int* __restrict__ pairs, int nk, int n,
               const __bf16* __restrict__ Xs, const __bf16* __restrict__ Wps,
               float* __restrict__ OUT) {
    constexpr int KB = CIN / 32;
    __shared__ u32x4 ldsB[KB * 512];                 // 16 KB (CIN=64) / 32 KB (CIN=128)
    const int t    = threadIdx.x;
    const int lane = t & 31;
    const int wave = t >> 5;
    const int h    = lane >> 4;
    const int lm   = lane & 15;
    const int m0   = blockIdx.x * 128 + wave * 16;
    const int r    = m0 + lm;

    v8f acc[8] = {};
    const u32x4* wpu = (const u32x4*)Wp;

    for (int k = 0; k < nk; ++k) {
        const int  idx   = (r < n) ? pairs[(size_t)k * n + r] : -1;
        const bool valid = idx >= 0;
        const __bf16* row = X + (size_t)(valid ? idx : 0) * CIN;

        // gather A operands for all K-blocks of this offset (overlaps staging)
        u32x4 alo[KB], ahi[KB];
#pragma unroll
        for (int kb = 0; kb < KB; ++kb) {
            alo[kb] = u32x4{0u, 0u, 0u, 0u};
            ahi[kb] = u32x4{0u, 0u, 0u, 0u};
            if (valid) {  // A layout: lane=row, K = kb*32 + 16h + {0..15}
                alo[kb] = *(const u32x4*)(row + kb * 32 + 16 * h);
                ahi[kb] = *(const u32x4*)(row + kb * 32 + 16 * h + 8);
            }
        }

        __syncthreads();                              // prior reads of ldsB done
        {
            const u32x4* src = wpu + (size_t)k * KB * 512;
#pragma unroll
            for (int i = 0; i < KB * 2; ++i)
                copy_chunk(&ldsB[i * 256 + t], &src[i * 256 + t]);
        }
        stage_fence();
        __syncthreads();                              // ldsB fully populated

#pragma unroll
        for (int kb = 0; kb < KB; ++kb)
            gemm_step(&ldsB[kb * 512], alo[kb], ahi[kb], lane, acc);
    }

    if constexpr (SKIP) {   // identity = feats @ W_skip (Cin=64), fused
        const u32x4* wsu = (const u32x4*)Wps;
        const bool valid = r < n;
        const __bf16* row = Xs + (size_t)(valid ? r : 0) * 64;

        u32x4 alo[2], ahi[2];
#pragma unroll
        for (int kb = 0; kb < 2; ++kb) {
            alo[kb] = u32x4{0u, 0u, 0u, 0u};
            ahi[kb] = u32x4{0u, 0u, 0u, 0u};
            if (valid) {
                alo[kb] = *(const u32x4*)(row + kb * 32 + 16 * h);
                ahi[kb] = *(const u32x4*)(row + kb * 32 + 16 * h + 8);
            }
        }

        __syncthreads();
#pragma unroll
        for (int i = 0; i < 4; ++i)
            copy_chunk(&ldsB[i * 256 + t], &wsu[i * 256 + t]);
        stage_fence();
        __syncthreads();

#pragma unroll
        for (int kb = 0; kb < 2; ++kb)
            gemm_step(&ldsB[kb * 512], alo[kb], ahi[kb], lane, acc);
    }

    // C/D layout: VGPR v -> M = v + 8*(lane>>4), N = nt*16 + (lane&15)
#pragma unroll
    for (int v = 0; v < 8; ++v) {
        const int m = m0 + v + 8 * h;
        if (m < n) {
            float* o = OUT + (size_t)m * 128 + lm;
#pragma unroll
            for (int nt = 0; nt < 8; ++nt) o[nt * 16] = acc[nt][v];
        }
    }
}

// ---------- launcher ----------
extern "C" void kernel_launch(void* const* d_in, const int* in_sizes, int n_in,
                              void* d_out, int out_size, void* d_ws, size_t ws_size,
                              hipStream_t stream) {
    (void)n_in; (void)out_size; (void)ws_size;
    const float* feats  = (const float*)d_in[0];
    const float* W1     = (const float*)d_in[1];
    const float* W2     = (const float*)d_in[2];
    const float* Wskip  = (const float*)d_in[3];
    const float* gamma1 = (const float*)d_in[4];
    const float* beta1  = (const float*)d_in[5];
    const float* gamma2 = (const float*)d_in[6];
    const float* beta2  = (const float*)d_in[7];
    const int*   pairs  = (const int*)d_in[8];

    const int N = in_sizes[0] / 64;   // IN_CH = 64

    // workspace carve (~206 MB)
    char* w = (char*)d_ws;
    size_t off = 0;
    auto carve = [&](size_t bytes) {
        void* p = w + off;
        off += (bytes + 255) & ~(size_t)255;
        return p;
    };
    float*  psum = (float*)carve(256 * 128 * sizeof(float));
    float*  psq  = (float*)carve(256 * 128 * sizeof(float));
    float*  a1   = (float*)carve(64 * sizeof(float));
    float*  bb1  = (float*)carve(64 * sizeof(float));
    float*  a2   = (float*)carve(128 * sizeof(float));
    float*  bb2  = (float*)carve(128 * sizeof(float));
    __bf16* x1   = (__bf16*)carve((size_t)N * 64 * 2);
    __bf16* fb   = (__bf16*)carve((size_t)N * 64 * 2);
    float*  h1   = (float*)carve((size_t)N * 128 * 4);
    __bf16* x2   = (__bf16*)carve((size_t)N * 128 * 2);
    __bf16* wp1  = (__bf16*)carve((size_t)9 * 64 * 128 * 2);
    __bf16* wp2  = (__bf16*)carve((size_t)9 * 128 * 128 * 2);
    __bf16* wps  = (__bf16*)carve((size_t)64 * 128 * 2);

    const int SB = 256;
    const float inv_n = 1.0f / (float)N;

    // BN1 -> relu -> bf16 (and raw feats bf16 for skip)
    bn_stats_partial<64><<<SB, 256, 0, stream>>>(feats, N, psum, psq);
    bn_params<<<1, 64, 0, stream>>>(psum, psq, gamma1, beta1, SB, 64, inv_n, a1, bb1);
    {
        int tot = N * 64;
        bn_relu_cast<<<(tot + 255) / 256, 256, 0, stream>>>(feats, a1, bb1, x1, fb, tot, 64);
    }

    // pack weights to WMMA-B layout
    pack_weights<<<(9 * 2 * 512 + 255) / 256, 256, 0, stream>>>(W1, wp1, 9, 2);
    pack_weights<<<(9 * 4 * 512 + 255) / 256, 256, 0, stream>>>(W2, wp2, 9, 4);
    pack_weights<<<(1 * 2 * 512 + 255) / 256, 256, 0, stream>>>(Wskip, wps, 1, 2);

    const int gb = (N + 127) / 128;

    // conv1 (64 -> 128)
    subm_conv_wmma<64, false><<<gb, 256, 0, stream>>>(x1, wp1, pairs, 9, N,
                                                      (const __bf16*)nullptr,
                                                      (const __bf16*)nullptr, h1);

    // BN2 -> relu -> bf16
    bn_stats_partial<128><<<SB, 256, 0, stream>>>(h1, N, psum, psq);
    bn_params<<<1, 128, 0, stream>>>(psum, psq, gamma2, beta2, SB, 128, inv_n, a2, bb2);
    {
        int tot = N * 128;
        bn_relu_cast<<<(tot + 255) / 256, 256, 0, stream>>>(h1, a2, bb2, x2,
                                                            (__bf16*)nullptr, tot, 128);
    }

    // conv2 (128 -> 128) + fused skip GEMM, writes d_out
    subm_conv_wmma<128, true><<<gb, 256, 0, stream>>>(x2, wp2, pairs, 9, N,
                                                      fb, wps, (float*)d_out);
}
